// LSTMContextualize_70136815943722
// MI455X (gfx1250) — matrix operand order repaired and checked by reading, not compiled, not run on hardware
//
#include <hip/hip_runtime.h>
#include <cstdint>
#include <cstddef>

#define BB 64
#define SS 2048
#define EE 512
#define HH 256
#define GG 1024   // 4*H
#define OO 512    // 2*H (output last dim)

typedef __attribute__((ext_vector_type(16))) __bf16 v16bf;
typedef __attribute__((ext_vector_type(8)))  __bf16 v8bf;
typedef __attribute__((ext_vector_type(8)))  float  v8f;
typedef __attribute__((ext_vector_type(4)))  int    v4i;

union ABfrag { v16bf v; v8bf h[2]; };

__device__ __forceinline__ unsigned short f2bf(float f) {
    return __builtin_bit_cast(unsigned short, (__bf16)f);
}
__device__ __forceinline__ float sigm_(float x) {
    return 1.f / (1.f + __expf(-x));
}
__device__ __forceinline__ float tanh_(float x) {
    x = fminf(fmaxf(x, -15.f), 15.f);
    float e = __expf(2.f * x);
    return (e - 1.f) / (e + 1.f);
}

// Async global->LDS 16-byte copy (ASYNCcnt-tracked, fire-and-forget).
// Builtin signature (from hipcc diagnostic): param0 = addrspace(1) v4i* src.
__device__ __forceinline__ void async_b128(void* lds_dst, const void* gsrc) {
#if __has_builtin(__builtin_amdgcn_global_load_async_to_lds_b128)
    __builtin_amdgcn_global_load_async_to_lds_b128(
        (__attribute__((address_space(1))) v4i*)(unsigned long long)gsrc,
        (__attribute__((address_space(3))) v4i*)(unsigned)(size_t)lds_dst,
        0, 0);
#else
    unsigned loff = (unsigned)(size_t)lds_dst;
    unsigned long long ga = (unsigned long long)(size_t)gsrc;
    asm volatile("global_load_async_to_lds_b128 %0, %1, off"
                 :: "v"(loff), "v"(ga) : "memory");
#endif
}
__device__ __forceinline__ void wait_async_le8() {
    asm volatile("s_wait_asynccnt 0x8" ::: "memory");
}
__device__ __forceinline__ void wait_async_0() {
    asm volatile("s_wait_asynccnt 0x0" ::: "memory");
}

// ---------------------------------------------------------------------------
// Pack context_emb [B,S,E] fp32 -> time-major [S,B,E] bf16 (one GEMM operand).
// ---------------------------------------------------------------------------
__global__ void pack_x(const float* __restrict__ x, unsigned short* __restrict__ xbf) {
    size_t tid = (size_t)blockIdx.x * blockDim.x + threadIdx.x;
    size_t total = (size_t)BB * SS * (EE / 8);
    if (tid >= total) return;
    unsigned e8 = tid % (EE / 8);
    unsigned b  = (tid / (EE / 8)) % BB;
    unsigned s  = tid / ((size_t)(EE / 8) * BB);
    const float4* src = (const float4*)(x + ((size_t)b * SS + s) * EE + e8 * 8);
    float4 f0 = src[0], f1 = src[1];
    uint4 o;
    o.x = ((unsigned)f2bf(f0.y) << 16) | f2bf(f0.x);
    o.y = ((unsigned)f2bf(f0.w) << 16) | f2bf(f0.z);
    o.z = ((unsigned)f2bf(f1.y) << 16) | f2bf(f1.x);
    o.w = ((unsigned)f2bf(f1.w) << 16) | f2bf(f1.z);
    *(uint4*)(xbf + ((size_t)s * BB + b) * EE + e8 * 8) = o;
}

// ---------------------------------------------------------------------------
// Convert weights to bf16, fuse biases (b_ih + b_hh).
// ---------------------------------------------------------------------------
__global__ void pack_w(const float* __restrict__ wihF, const float* __restrict__ whhF,
                       const float* __restrict__ bihF, const float* __restrict__ bhhF,
                       const float* __restrict__ wihB, const float* __restrict__ whhB,
                       const float* __restrict__ bihB, const float* __restrict__ bhhB,
                       unsigned short* __restrict__ wihFo, unsigned short* __restrict__ whhFo,
                       float* __restrict__ biasF,
                       unsigned short* __restrict__ wihBo, unsigned short* __restrict__ whhBo,
                       float* __restrict__ biasB) {
    unsigned stride = gridDim.x * blockDim.x;
    unsigned tid = blockIdx.x * blockDim.x + threadIdx.x;
    for (unsigned i = tid; i < GG * EE; i += stride) {
        wihFo[i] = f2bf(wihF[i]);
        wihBo[i] = f2bf(wihB[i]);
    }
    for (unsigned i = tid; i < GG * HH; i += stride) {
        whhFo[i] = f2bf(whhF[i]);
        whhBo[i] = f2bf(whhB[i]);
    }
    for (unsigned i = tid; i < GG; i += stride) {
        biasF[i] = bihF[i] + bhhF[i];
        biasB[i] = bihB[i] + bhhB[i];
    }
}

// ---------------------------------------------------------------------------
// Input projection GEMM: gx[m, g] = sum_k xbf[m,k] * wih[g,k] + bias[g]
// M = S*B = 131072, N = 4H = 1024, K = E = 512.  bf16 WMMA, fp32 accumulate.
// ---------------------------------------------------------------------------
__global__ __launch_bounds__(256) void input_gemm(
    const unsigned short* __restrict__ xbf,
    const unsigned short* __restrict__ wihF, const unsigned short* __restrict__ wihB,
    const float* __restrict__ biasF, const float* __restrict__ biasB,
    float* __restrict__ gxF, float* __restrict__ gxB) {
    const unsigned short* wih = blockIdx.z ? wihB : wihF;
    const float* bias = blockIdx.z ? biasB : biasF;
    float* gx = blockIdx.z ? gxB : gxF;

    unsigned lane = threadIdx.x & 31;
    unsigned wv   = threadIdx.x >> 5;        // 0..7
    unsigned ln15 = lane & 15;
    bool hiH = lane >= 16;
    unsigned mbase = hiH ? 8u : 0u;

    size_t   r0 = (size_t)blockIdx.y * 128 + (wv & 3) * 32;   // 2 m-tiles
    unsigned c0 = blockIdx.x * 128 + (wv >> 2) * 64;          // 4 n-tiles

    v8f acc[2][4];
#pragma unroll
    for (int ni = 0; ni < 4; ++ni) {
        float bv = bias[c0 + ni * 16 + ln15];
#pragma unroll
        for (int mi = 0; mi < 2; ++mi)
#pragma unroll
            for (int v = 0; v < 8; ++v) acc[mi][ni][v] = bv;
    }

    const __bf16* X = (const __bf16*)xbf;
    const __bf16* W = (const __bf16*)wih;

    for (int kc = 0; kc < EE / 32; ++kc) {
        ABfrag a[2];
#pragma unroll
        for (int mi = 0; mi < 2; ++mi) {
            // ISA 16-bit A layout: lanes 0-15 hold K 0..7 / 16..23,
            // lanes 16-31 hold K 8..15 / 24..31 of the 32-wide chunk.
            const __bf16* ap = X + (r0 + mi * 16 + ln15) * EE + kc * 32 + (hiH ? 8 : 0);
            a[mi].h[0] = *(const v8bf*)(ap);
            a[mi].h[1] = *(const v8bf*)(ap + 16);
        }
#pragma unroll
        for (int ni = 0; ni < 4; ++ni) {
            // B[k][n] = wih[n][k]: column n's K values are contiguous.
            const __bf16* bp = W + (size_t)(c0 + ni * 16 + ln15) * EE + kc * 32 + (hiH ? 8 : 0);
            ABfrag bf_;
            bf_.h[0] = *(const v8bf*)bp;
            bf_.h[1] = *(const v8bf*)(bp + 16);
#pragma unroll
            for (int mi = 0; mi < 2; ++mi)
                acc[mi][ni] = __builtin_amdgcn_wmma_f32_16x16x32_bf16(
                    false, a[mi].v, false, bf_.v, (short)0, acc[mi][ni], false, false);
        }
    }
#pragma unroll
    for (int mi = 0; mi < 2; ++mi)
#pragma unroll
        for (int ni = 0; ni < 4; ++ni) {
            float* op = gx + (r0 + mi * 16 + mbase) * GG + c0 + ni * 16 + ln15;
#pragma unroll
            for (int v = 0; v < 8; ++v) op[(size_t)v * GG] = acc[mi][ni][v];
        }
}

// ---------------------------------------------------------------------------
// LSTM scan. Batch rows are recurrence-independent: grid = (4 batch tiles of
// 16 rows) x (2 directions), 16 waves each, no inter-WG traffic. Wave w owns
// h-columns [16w,16w+16) and the 4 matched i/f/g/o gate tiles -> the cell
// update is in-register; c lives in VGPRs across all 2048 steps.
// gx[t] is double-buffered in LDS via async global->LDS B128 copies: each
// wave stages exactly the 4KB slice it will consume next step, overlapped
// with the current step's WMMAs, completion via per-wave s_wait_asynccnt.
// ---------------------------------------------------------------------------
__global__ __launch_bounds__(512) void lstm_scan(
    const float* __restrict__ gxF, const float* __restrict__ gxB,
    const unsigned short* __restrict__ whhF, const unsigned short* __restrict__ whhB,
    const float* __restrict__ h0F, const float* __restrict__ c0F,
    const float* __restrict__ h0B, const float* __restrict__ c0B,
    float* __restrict__ out) {
    const int dir = blockIdx.y;                       // 0 = fwd, 1 = bwd
    const unsigned wgR0 = blockIdx.x * 16;            // this WG's batch rows
    const float* gx = dir ? gxB : gxF;
    const unsigned short* whh = dir ? whhB : whhF;
    const float* h0 = dir ? h0B : h0F;
    const float* c0 = dir ? c0B : c0F;

    // padded strides (68 floats / 264 bf16) => conflict-free ds access
    __shared__ __align__(16) float  gxs[2][16][16][68];   // [buf][wave][row][col]
    __shared__ __align__(16) __bf16 hbuf[16][264];

    unsigned lane = threadIdx.x & 31;
    unsigned wv   = threadIdx.x >> 5;                 // 0..15 : h-column group
    unsigned ln15 = lane & 15;
    bool hiH = lane >= 16;
    unsigned mbase = hiH ? 8u : 0u;
    unsigned hc0 = wv * 16;                           // this wave's h columns

    // initial h into LDS (bf16)
    for (unsigned i = threadIdx.x; i < 16 * HH; i += blockDim.x) {
        unsigned r = i >> 8, c = i & 255;
        hbuf[r][c] = (__bf16)h0[(size_t)(wgR0 + r) * HH + c];
    }
    // initial c fragment (stays in registers for the whole scan)
    v8f cfrag;
#pragma unroll
    for (int v = 0; v < 8; ++v)
        cfrag[v] = c0[(size_t)(wgR0 + mbase + v) * HH + hc0 + ln15];

    // stage gx[t] slice for this wave: 16 rows x 64 gate cols (4 tiles)
    auto stage = [&](int buf, int tt) {
#pragma unroll
        for (int j = 0; j < 8; ++j) {
            unsigned chunk = lane + 32 * j;           // 256 chunks of 4 floats
            unsigned row = chunk >> 4;                // 0..15
            unsigned cg  = chunk & 15;                // col-group of 4
            unsigned q   = cg >> 2;                   // gate
            unsigned cc  = (cg & 3) * 4;              // col within tile
            const float* gp = gx + ((size_t)tt * BB + wgR0 + row) * GG
                            + q * HH + hc0 + cc;
            async_b128(&gxs[buf][wv][row][q * 16 + cc], gp);
        }
    };
    int t0 = dir ? (SS - 1) : 0;
    stage(0, t0);
    __syncthreads();

    const __bf16* W = (const __bf16*)whh;
    int buf = 0;

    for (int step = 0; step < SS; ++step) {
        int t = dir ? (SS - 1 - step) : step;
        bool more = (step + 1 < SS);
        if (more) {
            int tn = dir ? (SS - 2 - step) : step + 1;
            stage(buf ^ 1, tn);                       // overlap with this step
            wait_async_le8();                         // current buffer landed
        } else {
            wait_async_0();
        }

        // init accumulators from staged input projection gx[t]
        v8f acc[4];
#pragma unroll
        for (int q = 0; q < 4; ++q) {
            const float* sp = &gxs[buf][wv][mbase][q * 16 + ln15];
#pragma unroll
            for (int v = 0; v < 8; ++v) acc[q][v] = sp[(size_t)v * 68];
        }

        // acc += h @ w_hh^T  (K = 256, 8 chunks of 32)
        const __bf16* hrow = &hbuf[ln15][0];
        for (int kc = 0; kc < HH / 32; ++kc) {
            ABfrag a;
            const __bf16* ap = hrow + kc * 32 + (hiH ? 8 : 0);
            a.h[0] = *(const v8bf*)ap;
            a.h[1] = *(const v8bf*)(ap + 16);
#pragma unroll
            for (int q = 0; q < 4; ++q) {
                const __bf16* bp = W + (size_t)(q * HH + hc0 + ln15) * HH
                                 + kc * 32 + (hiH ? 8 : 0);
                ABfrag bfr;
                bfr.h[0] = *(const v8bf*)bp;
                bfr.h[1] = *(const v8bf*)(bp + 16);
                acc[q] = __builtin_amdgcn_wmma_f32_16x16x32_bf16(
                    false, a.v, false, bfr.v, (short)0, acc[q], false, false);
            }
        }

        // gate math in-register (PyTorch order i,f,g,o); emit output
        float hv[8];
        float* op = out + ((size_t)(wgR0 + mbase) * SS + t) * OO + dir * HH + hc0 + ln15;
#pragma unroll
        for (int v = 0; v < 8; ++v) {
            float ig = sigm_(acc[0][v]);
            float fg = sigm_(acc[1][v]);
            float gg = tanh_(acc[2][v]);
            float og = sigm_(acc[3][v]);
            float cn = fg * cfrag[v] + ig * gg;
            cfrag[v] = cn;
            float hn = og * tanh_(cn);
            hv[v] = hn;
            op[(size_t)v * SS * OO] = hn;
        }
        __syncthreads();   // all waves done reading hbuf for this step
#pragma unroll
        for (int v = 0; v < 8; ++v)
            hbuf[mbase + v][hc0 + ln15] = (__bf16)hv[v];
        __syncthreads();   // new h visible before next step's reads
        buf ^= 1;
    }
}

extern "C" void kernel_launch(void* const* d_in, const int* in_sizes, int n_in,
                              void* d_out, int out_size, void* d_ws, size_t ws_size,
                              hipStream_t stream) {
    const float* x     = (const float*)d_in[0];
    const float* wihF  = (const float*)d_in[1];
    const float* whhF  = (const float*)d_in[2];
    const float* bihF  = (const float*)d_in[3];
    const float* bhhF  = (const float*)d_in[4];
    const float* wihB  = (const float*)d_in[5];
    const float* whhB  = (const float*)d_in[6];
    const float* bihB  = (const float*)d_in[7];
    const float* bhhB  = (const float*)d_in[8];
    const float* h0F   = (const float*)d_in[9];
    const float* c0F   = (const float*)d_in[10];
    const float* h0B   = (const float*)d_in[11];
    const float* c0B   = (const float*)d_in[12];
    float* out = (float*)d_out;

    // workspace carve-up
    char* p = (char*)d_ws;
    unsigned short* xbf   = (unsigned short*)p; p += (size_t)SS * BB * EE * 2;  // 128 MB
    unsigned short* wihFo = (unsigned short*)p; p += (size_t)GG * EE * 2;       // 1 MB
    unsigned short* wihBo = (unsigned short*)p; p += (size_t)GG * EE * 2;
    unsigned short* whhFo = (unsigned short*)p; p += (size_t)GG * HH * 2;       // 512 KB
    unsigned short* whhBo = (unsigned short*)p; p += (size_t)GG * HH * 2;
    float* biasF = (float*)p; p += (size_t)GG * 4;
    float* biasB = (float*)p; p += (size_t)GG * 4;
    float* gxF   = (float*)p; p += (size_t)SS * BB * GG * 4;                    // 512 MB
    float* gxB   = (float*)p; p += (size_t)SS * BB * GG * 4;                    // 512 MB
    (void)ws_size; (void)in_sizes; (void)n_in; (void)out_size;

    {   // pack x: one thread per 8 elems
        size_t total = (size_t)BB * SS * (EE / 8);
        pack_x<<<(unsigned)((total + 255) / 256), 256, 0, stream>>>(x, xbf);
    }
    pack_w<<<1024, 256, 0, stream>>>(wihF, whhF, bihF, bhhF, wihB, whhB, bihB, bhhB,
                                     wihFo, whhFo, biasF, wihBo, whhBo, biasB);
    {   // M/128 = 1024 row blocks, N/128 = 8 col blocks, 2 directions
        dim3 grid(GG / 128, (SS * BB) / 128, 2);
        input_gemm<<<grid, 256, 0, stream>>>(xbf, wihFo, wihBo, biasF, biasB, gxF, gxB);
    }
    lstm_scan<<<dim3(4, 2), 512, 0, stream>>>(gxF, gxB, whhFo, whhBo,
                                              h0F, c0F, h0B, c0B, out);
}
